// GATClassifier_36859409334831
// MI455X (gfx1250) — matrix operand order repaired and checked by reading, not compiled, not run on hardware
//
#include <hip/hip_runtime.h>

#define N_NODES 50000
#define N_EDGES 800000
#define EP (N_EDGES + N_NODES)   // edges + self-loops = 850000
#define IN_C 128
#define HID 32
#define HEADS 8
#define F1 (HEADS * HID)         // 256
#define NCLS 10
#define NEG_SLOPE 0.2f

typedef __attribute__((ext_vector_type(16))) _Float16 v16h;
typedef __attribute__((ext_vector_type(8)))  float    v8f;

// ---- order-preserving float <-> uint for atomicMax-based segment max ----
__device__ __forceinline__ unsigned f2ord(float f) {
    unsigned u = __float_as_uint(f);
    return (u & 0x80000000u) ? ~u : (u | 0x80000000u);
}
__device__ __forceinline__ float ord2f(unsigned o) {
    return (o & 0x80000000u) ? __uint_as_float(o ^ 0x80000000u)
                             : __uint_as_float(~o);
}

// ================= GEMM1: h1 = X(50000x128) @ W1(128x256), f16 WMMA =======
// W1 staged once per workgroup into LDS as f16 (64 KB of the 320 KB WGP LDS);
// B fragments then come from ds_load, A fragments from coalesced global loads.
__global__ void gemm1_wmma(const float* __restrict__ X,
                           const float* __restrict__ W,
                           float* __restrict__ H) {
    __shared__ _Float16 wlds[IN_C * F1];          // 128*256 f16 = 64 KB
    int tid = threadIdx.x;
    for (int i = tid; i < IN_C * F1; i += 256)    // coalesced global, f32->f16
        wlds[i] = (_Float16)W[i];
    __syncthreads();

    int wave = (int)((blockIdx.x * blockDim.x + tid) >> 5);
    int lane = tid & 31;
    int g = lane >> 4, r = lane & 15;
    // 3125 x 16 tiles of 16x16
    int mt = wave >> 4;
    int nt = wave & 15;
    if (mt >= 3125) return;            // wave-uniform guard (EXEC stays full)
    int row  = mt * 16 + r;
    int ncol = nt * 16 + r;
    v8f acc = {};
    for (int kb = 0; kb < IN_C; kb += 32) {
        v16h a, b;
        // A fragment (16-bit 16x32): lane holds K = kb+8g+j (j<8), kb+16+8g+(j-8)
        const float* ap1 = X + (size_t)row * IN_C + kb + 8 * g;
        const float* ap2 = ap1 + 16;
#pragma unroll
        for (int j = 0; j < 8; ++j) {
            a[j]     = (_Float16)ap1[j];
            a[8 + j] = (_Float16)ap2[j];
        }
        // B fragment from LDS: lane group g holds K = kb+8g+j (j<8), kb+16+8g+(j-8), column = ncol
#pragma unroll
        for (int j = 0; j < 16; ++j)
            b[j] = wlds[(kb + 8 * g + j + 8 * (j >> 3)) * F1 + ncol];
        acc = __builtin_amdgcn_wmma_f32_16x16x32_f16(false, a, false, b,
                                                     (short)0, acc, false, false);
    }
#pragma unroll
    for (int i = 0; i < 8; ++i)
        H[(size_t)(mt * 16 + i + 8 * g) * F1 + ncol] = acc[i];
}

// ================= GEMM2: h2 = Hact(50000x256) @ W2(256x10->pad16) ========
__global__ void gemm2_wmma(const float* __restrict__ Hin,
                           const float* __restrict__ W,
                           float* __restrict__ H2) {
    __shared__ _Float16 w2[F1 * 16];              // 256*16 f16 = 8 KB, zero-padded
    int tid = threadIdx.x;
    for (int i = tid; i < F1 * 16; i += 256) {
        int k = i >> 4, c = i & 15;
        w2[i] = (c < NCLS) ? (_Float16)W[(size_t)k * NCLS + c] : (_Float16)0.0f;
    }
    __syncthreads();

    int wave = (int)((blockIdx.x * blockDim.x + tid) >> 5);
    int lane = tid & 31;
    int g = lane >> 4, r = lane & 15;
    if (wave >= 3125) return;          // wave-uniform guard
    int row = wave * 16 + r;
    v8f acc = {};
    for (int kb = 0; kb < F1; kb += 32) {
        v16h a, b;
        const float* ap1 = Hin + (size_t)row * F1 + kb + 8 * g;
        const float* ap2 = ap1 + 16;
#pragma unroll
        for (int j = 0; j < 8; ++j) {
            a[j]     = (_Float16)ap1[j];
            a[8 + j] = (_Float16)ap2[j];
        }
#pragma unroll
        for (int j = 0; j < 16; ++j)
            b[j] = w2[(kb + 8 * g + j + 8 * (j >> 3)) * 16 + r];
        acc = __builtin_amdgcn_wmma_f32_16x16x32_f16(false, a, false, b,
                                                     (short)0, acc, false, false);
    }
#pragma unroll
    for (int i = 0; i < 8; ++i)
        H2[(size_t)(wave * 16 + i + 8 * g) * 16 + r] = acc[i];
}

// ================= attention coefficients, layer 1 ========================
__global__ void attn1(const float* __restrict__ H,
                      const float* __restrict__ ws, const float* __restrict__ wd,
                      float* __restrict__ a_s, float* __restrict__ a_d) {
    int idx = blockIdx.x * blockDim.x + threadIdx.x;   // node*8 + head
    if (idx >= N_NODES * HEADS) return;
    int n = idx >> 3, h = idx & 7;
    const float* hp = H + (size_t)n * F1 + h * HID;
    const float* sp = ws + h * HID;
    const float* dp = wd + h * HID;
    float s = 0.f, d = 0.f;
#pragma unroll
    for (int c = 0; c < HID; ++c) {
        float v = hp[c];
        s += v * sp[c];
        d += v * dp[c];
    }
    a_s[idx] = s;
    a_d[idx] = d;
}

// ================= per-edge logits + segment max (layer 1) ================
__global__ void scores1(const int* __restrict__ ei,
                        const float* __restrict__ a_s, const float* __restrict__ a_d,
                        float* __restrict__ logit, unsigned* __restrict__ nmax) {
    int idx = blockIdx.x * blockDim.x + threadIdx.x;
    if (idx >= EP * HEADS) return;
    int e = idx >> 3, h = idx & 7;
    int s, d;
    if (e < N_EDGES) { s = ei[e]; d = ei[N_EDGES + e]; } else { s = d = e - N_EDGES; }
    float v = a_s[s * HEADS + h] + a_d[d * HEADS + h];
    v = (v > 0.f) ? v : NEG_SLOPE * v;
    logit[idx] = v;
    atomicMax(&nmax[d * HEADS + h], f2ord(v));
}

// ================= exp + segment sum (layer 1) ============================
__global__ void expsum1(const int* __restrict__ ei, const unsigned* __restrict__ nmax,
                        float* __restrict__ wbuf, float* __restrict__ denom) {
    int idx = blockIdx.x * blockDim.x + threadIdx.x;
    if (idx >= EP * HEADS) return;
    int e = idx >> 3, h = idx & 7;
    int d = (e < N_EDGES) ? ei[N_EDGES + e] : e - N_EDGES;
    float m = ord2f(nmax[d * HEADS + h]);
    float w = expf(wbuf[idx] - m);
    wbuf[idx] = w;
    atomicAdd(&denom[d * HEADS + h], w);
}

// ====== weighted aggregation (layer 1): persistent blocks, grid-stride ====
// h1 / h_act are L2-resident (51 MB each vs 192 MB L2); prefetch the next
// edge's source row (global_prefetch_b8) to hide L2 latency across iters.
__global__ void aggregate1(const int* __restrict__ ei, const float* __restrict__ H,
                           const float* __restrict__ wbuf, const float* __restrict__ denom,
                           float* __restrict__ out) {
    int t = threadIdx.x;        // 0..255 = head*32 + chan
    int h = t >> 5;
    for (int e = blockIdx.x; e < EP; e += gridDim.x) {
        int s, d;
        if (e < N_EDGES) { s = ei[e]; d = ei[N_EDGES + e]; } else { s = d = e - N_EDGES; }
        int en = e + gridDim.x;
        if (en < EP) {
            int sn = (en < N_EDGES) ? ei[en] : en - N_EDGES;
            __builtin_prefetch(&H[(size_t)sn * F1 + t], 0, 3);
        }
        float alpha = wbuf[e * HEADS + h] / denom[d * HEADS + h];
        atomicAdd(&out[(size_t)d * F1 + t], alpha * H[(size_t)s * F1 + t]);
    }
}

// ================= bias + ELU (in place) ==================================
__global__ void elu_bias(const float* __restrict__ b, float* __restrict__ buf) {
    int idx = blockIdx.x * blockDim.x + threadIdx.x;
    if (idx >= N_NODES * F1) return;
    float v = buf[idx] + b[idx & (F1 - 1)];
    buf[idx] = (v > 0.f) ? v : (expf(v) - 1.f);
}

// ================= attention coefficients, layer 2 (1 head) ===============
__global__ void attn2(const float* __restrict__ H2,
                      const float* __restrict__ ws, const float* __restrict__ wd,
                      float* __restrict__ a_s, float* __restrict__ a_d) {
    int n = blockIdx.x * blockDim.x + threadIdx.x;
    if (n >= N_NODES) return;
    float s = 0.f, d = 0.f;
#pragma unroll
    for (int c = 0; c < NCLS; ++c) {
        float v = H2[(size_t)n * 16 + c];
        s += v * ws[c];
        d += v * wd[c];
    }
    a_s[n] = s;
    a_d[n] = d;
}

__global__ void scores2(const int* __restrict__ ei,
                        const float* __restrict__ a_s, const float* __restrict__ a_d,
                        float* __restrict__ logit, unsigned* __restrict__ nmax) {
    int e = blockIdx.x * blockDim.x + threadIdx.x;
    if (e >= EP) return;
    int s, d;
    if (e < N_EDGES) { s = ei[e]; d = ei[N_EDGES + e]; } else { s = d = e - N_EDGES; }
    float v = a_s[s] + a_d[d];
    v = (v > 0.f) ? v : NEG_SLOPE * v;
    logit[e] = v;
    atomicMax(&nmax[d], f2ord(v));
}

__global__ void expsum2(const int* __restrict__ ei, const unsigned* __restrict__ nmax,
                        float* __restrict__ wbuf, float* __restrict__ denom) {
    int e = blockIdx.x * blockDim.x + threadIdx.x;
    if (e >= EP) return;
    int d = (e < N_EDGES) ? ei[N_EDGES + e] : e - N_EDGES;
    float w = expf(wbuf[e] - ord2f(nmax[d]));
    wbuf[e] = w;
    atomicAdd(&denom[d], w);
}

__global__ void aggregate2(const int* __restrict__ ei, const float* __restrict__ H2,
                           const float* __restrict__ wbuf, const float* __restrict__ denom,
                           float* __restrict__ out) {
    int stride = gridDim.x * blockDim.x;
    for (int idx = blockIdx.x * blockDim.x + threadIdx.x; idx < EP * NCLS; idx += stride) {
        int e = idx / NCLS, c = idx - e * NCLS;
        int s, d;
        if (e < N_EDGES) { s = ei[e]; d = ei[N_EDGES + e]; } else { s = d = e - N_EDGES; }
        float alpha = wbuf[e] / denom[d];
        atomicAdd(&out[(size_t)d * NCLS + c], alpha * H2[(size_t)s * 16 + c]);
    }
}

// ================= bias + log_softmax =====================================
__global__ void logsm(const float* __restrict__ agg, const float* __restrict__ b,
                      float* __restrict__ out) {
    int n = blockIdx.x * blockDim.x + threadIdx.x;
    if (n >= N_NODES) return;
    float v[NCLS];
    float m = -3.4e38f;
#pragma unroll
    for (int c = 0; c < NCLS; ++c) {
        v[c] = agg[(size_t)n * NCLS + c] + b[c];
        m = fmaxf(m, v[c]);
    }
    float se = 0.f;
#pragma unroll
    for (int c = 0; c < NCLS; ++c) se += expf(v[c] - m);
    float lse = logf(se) + m;
#pragma unroll
    for (int c = 0; c < NCLS; ++c) out[(size_t)n * NCLS + c] = v[c] - lse;
}

extern "C" void kernel_launch(void* const* d_in, const int* in_sizes, int n_in,
                              void* d_out, int out_size, void* d_ws, size_t ws_size,
                              hipStream_t stream) {
    (void)in_sizes; (void)n_in; (void)out_size; (void)ws_size;
    const float* x   = (const float*)d_in[0];
    const int*   ei  = (const int*)  d_in[1];
    const float* W1  = (const float*)d_in[2];
    const float* as1 = (const float*)d_in[3];
    const float* ad1 = (const float*)d_in[4];
    const float* b1  = (const float*)d_in[5];
    const float* W2  = (const float*)d_in[6];
    const float* as2 = (const float*)d_in[7];
    const float* ad2 = (const float*)d_in[8];
    const float* b2  = (const float*)d_in[9];
    float* out = (float*)d_out;

    char* ws = (char*)d_ws;
    size_t off = 0;
    auto alloc = [&](size_t bytes) -> void* {
        void* p = ws + off;
        off += (bytes + 255) & ~(size_t)255;
        return p;
    };
    float*    h1    = (float*)   alloc((size_t)N_NODES * F1 * 4);
    float*    hact  = (float*)   alloc((size_t)N_NODES * F1 * 4);
    float*    a_s1  = (float*)   alloc((size_t)N_NODES * HEADS * 4);
    float*    a_d1  = (float*)   alloc((size_t)N_NODES * HEADS * 4);
    unsigned* nmax1 = (unsigned*)alloc((size_t)N_NODES * HEADS * 4);
    float*    den1  = (float*)   alloc((size_t)N_NODES * HEADS * 4);
    float*    wbuf  = (float*)   alloc((size_t)EP * HEADS * 4);
    float*    h2    = (float*)   alloc((size_t)N_NODES * 16 * 4);
    float*    a_s2  = (float*)   alloc((size_t)N_NODES * 4);
    float*    a_d2  = (float*)   alloc((size_t)N_NODES * 4);
    unsigned* nmax2 = (unsigned*)alloc((size_t)N_NODES * 4);
    float*    den2  = (float*)   alloc((size_t)N_NODES * 4);
    float*    agg2  = (float*)   alloc((size_t)N_NODES * NCLS * 4);

    hipMemsetAsync(hact,  0, (size_t)N_NODES * F1 * 4,    stream);
    hipMemsetAsync(nmax1, 0, (size_t)N_NODES * HEADS * 4, stream);
    hipMemsetAsync(den1,  0, (size_t)N_NODES * HEADS * 4, stream);
    hipMemsetAsync(nmax2, 0, (size_t)N_NODES * 4,         stream);
    hipMemsetAsync(den2,  0, (size_t)N_NODES * 4,         stream);
    hipMemsetAsync(agg2,  0, (size_t)N_NODES * NCLS * 4,  stream);

    // Layer 1
    gemm1_wmma<<<6250, 256, 0, stream>>>(x, W1, h1);                    // 50000 tiles / 8 waves
    attn1<<<(N_NODES * HEADS + 255) / 256, 256, 0, stream>>>(h1, as1, ad1, a_s1, a_d1);
    scores1<<<(EP * HEADS + 255) / 256, 256, 0, stream>>>(ei, a_s1, a_d1, wbuf, nmax1);
    expsum1<<<(EP * HEADS + 255) / 256, 256, 0, stream>>>(ei, nmax1, wbuf, den1);
    aggregate1<<<8192, 256, 0, stream>>>(ei, h1, wbuf, den1, hact);
    elu_bias<<<(N_NODES * F1 + 255) / 256, 256, 0, stream>>>(b1, hact);

    // Layer 2
    gemm2_wmma<<<(3125 + 7) / 8, 256, 0, stream>>>(hact, W2, h2);
    attn2<<<(N_NODES + 255) / 256, 256, 0, stream>>>(h2, as2, ad2, a_s2, a_d2);
    scores2<<<(EP + 255) / 256, 256, 0, stream>>>(ei, a_s2, a_d2, wbuf, nmax2);
    expsum2<<<(EP + 255) / 256, 256, 0, stream>>>(ei, nmax2, wbuf, den2);
    aggregate2<<<4096, 256, 0, stream>>>(ei, h2, wbuf, den2, agg2);
    logsm<<<(N_NODES + 255) / 256, 256, 0, stream>>>(agg2, b2, out);
}
